// SequencetoSequence_48412871360895
// MI455X (gfx1250) — matrix operand order repaired and checked by reading, not compiled, not run on hardware
//
#include <hip/hip_runtime.h>
#include <hip/hip_bf16.h>

typedef __attribute__((ext_vector_type(16))) _Float16 v16h;
typedef __attribute__((ext_vector_type(8)))  float    v8f;

#define V_   32000
#define E_   200
#define EP_  224      // E padded to multiple of 32 for WMMA K
#define H_   512
#define G4H_ 2048     // 4*H
#define S_   64
#define T_   64
#define B_   64

// ---------------- WMMA tile loaders (16x16x32 f16, wave32 layouts) ----------

// A tile: 16x32 f16.  lane L holds row m=L&15; lanes<16 hold K in {kb..kb+7, kb+16..kb+23},
// lanes>=16 hold K in {kb+8..kb+15, kb+24..kb+31}.  Two 16B loads per lane.
__device__ inline v16h load_tile_a(const _Float16* __restrict__ p, int ld, int kb) {
  int lane = threadIdx.x & 31;
  int m = lane & 15;
  int ko = kb + ((lane & 16) ? 8 : 0);
  const _Float16* row = p + m * ld + ko;
  union { v16h v; uint4 q[2]; } u;
  u.q[0] = *(const uint4*)(row);
  u.q[1] = *(const uint4*)(row + 16);
  return u.v;
}

// B tile: 32x16 f16 = W.T tile, loaded straight from row-major W[j,k].
// lane L holds column n=L&15 (= W row n); lanes<16 hold K in {kb..kb+7, kb+16..kb+23}, lanes>=16 K in {kb+8..kb+15, kb+24..kb+31}.
__device__ inline v16h load_tile_b(const _Float16* __restrict__ p, int ld, int kb) {
  int lane = threadIdx.x & 31;
  int n = lane & 15;
  int ko = kb + ((lane & 16) ? 8 : 0);
  const _Float16* row = p + n * ld + ko;
  union { v16h v; uint4 q[2]; } u;
  u.q[0] = *(const uint4*)(row);
  u.q[1] = *(const uint4*)(row + 16);
  return u.v;
}

#define WMMA_F16(a, b, c) \
  __builtin_amdgcn_wmma_f32_16x16x32_f16(false, (a), false, (b), (short)0, (c), false, false)

// ---------------- GEMM kernels ----------------------------------------------

// gates[64,2048] = X[64,kx] @ Wih.T + Hs[64,512] @ Whh.T   (f16 in, f32 accum)
// 512 tiles of 16x16; one wave per tile (parallelism > reuse: latency-bound step).
__global__ void lstm_gates_wmma(const _Float16* __restrict__ X, int kx,
                                const _Float16* __restrict__ Wih,
                                const _Float16* __restrict__ Hs,
                                const _Float16* __restrict__ Whh,
                                float* __restrict__ gates) {
  int wave = blockIdx.x * (blockDim.x >> 5) + (threadIdx.x >> 5);
  int tn = wave & 127;   // 128 tiles over 2048
  int tm = wave >> 7;    // 4 tiles over 64
  int lane = threadIdx.x & 31;
  v8f c = {};
  const _Float16* Xa = X   + (size_t)(tm * 16) * kx;
  const _Float16* Wb = Wih + (size_t)(tn * 16) * kx;
  for (int k = 0; k < kx; k += 32) {
    v16h a = load_tile_a(Xa, kx, k);
    v16h b = load_tile_b(Wb, kx, k);
    c = WMMA_F16(a, b, c);
  }
  const _Float16* Ha  = Hs  + (size_t)(tm * 16) * H_;
  const _Float16* Wb2 = Whh + (size_t)(tn * 16) * H_;
  for (int k = 0; k < H_; k += 32) {
    v16h a = load_tile_a(Ha, H_, k);
    v16h b = load_tile_b(Wb2, H_, k);
    c = WMMA_F16(a, b, c);
  }
  int m0 = (lane & 16) ? 8 : 0;
  int n  = lane & 15;
#pragma unroll
  for (int r = 0; r < 8; ++r)
    gates[(size_t)(tm * 16 + m0 + r) * G4H_ + tn * 16 + n] = c[r];
}

// out[64,32000] = H1[64,512] @ fcW.T + fc_b ; fcW16 row-major [32000,512] f16.
// 1x4 N-blocking: each wave computes 4 adjacent 16x16 tiles -> 4 wmma per 10 b128,
// A fragment amortized 4x.  2000 waves = 250 blocks x 8 waves.
__global__ void fc_wmma(const _Float16* __restrict__ H1,
                        const _Float16* __restrict__ W,
                        const float* __restrict__ bias,
                        float* __restrict__ out) {
  int wave = blockIdx.x * (blockDim.x >> 5) + (threadIdx.x >> 5);
  int tng = wave % 500;        // 500 groups of 4 N-tiles (2000 N tiles total)
  int tm  = wave / 500;        // 4 M tiles
  int tn0 = tng * 4;
  int lane = threadIdx.x & 31;
  v8f c0 = {}, c1 = {}, c2 = {}, c3 = {};
  const _Float16* Xa = H1 + (size_t)(tm * 16) * H_;
  const _Float16* Wb = W  + (size_t)(tn0 * 16) * H_;
#pragma unroll 2
  for (int k = 0; k < H_; k += 32) {
    v16h a  = load_tile_a(Xa, H_, k);
    v16h b0 = load_tile_b(Wb,            H_, k);
    v16h b1 = load_tile_b(Wb + 16 * H_,  H_, k);
    v16h b2 = load_tile_b(Wb + 32 * H_,  H_, k);
    v16h b3 = load_tile_b(Wb + 48 * H_,  H_, k);
    c0 = WMMA_F16(a, b0, c0);
    c1 = WMMA_F16(a, b1, c1);
    c2 = WMMA_F16(a, b2, c2);
    c3 = WMMA_F16(a, b3, c3);
  }
  int m0 = (lane & 16) ? 8 : 0;
  int nl = lane & 15;
  float* orow = out + (size_t)(tm * 16 + m0) * V_ + tn0 * 16 + nl;
  float bb0 = bias[tn0 * 16 + nl];
  float bb1 = bias[tn0 * 16 + 16 + nl];
  float bb2 = bias[tn0 * 16 + 32 + nl];
  float bb3 = bias[tn0 * 16 + 48 + nl];
#pragma unroll
  for (int r = 0; r < 8; ++r) {
    float* p = orow + (size_t)r * V_;
    p[0]  = c0[r] + bb0;
    p[16] = c1[r] + bb1;
    p[32] = c2[r] + bb2;
    p[48] = c3[r] + bb3;
  }
}

// ---------------- pointwise / helper kernels --------------------------------

__device__ inline float sigf(float x) { return 1.0f / (1.0f + __expf(-x)); }

// PyTorch gate order i,f,g,o over 4H; update c (f32) and h (stored as f16).
__global__ void lstm_update(const float* __restrict__ gates,
                            const float* __restrict__ bias,
                            float* __restrict__ c,
                            _Float16* __restrict__ h16) {
  int idx = blockIdx.x * blockDim.x + threadIdx.x;     // 0 .. B*H
  if (idx >= B_ * H_) return;
  int b = idx >> 9;          // /512
  int n = idx & 511;
  const float* g = gates + (size_t)b * G4H_;
  float gi = sigf (g[n]          + bias[n]);
  float gf = sigf (g[512 + n]    + bias[512 + n]);
  float gg = tanhf(g[1024 + n]   + bias[1024 + n]);
  float go = sigf (g[1536 + n]   + bias[1536 + n]);
  float cn = gf * c[idx] + gi * gg;
  c[idx] = cn;
  h16[idx] = (_Float16)(go * tanhf(cn));
}

// Gather B embedding rows -> f16, zero-padded to EP_ columns.
__global__ void embed_gather(const int* __restrict__ toks,
                             const float* __restrict__ table,  // [V,E]
                             _Float16* __restrict__ X16) {     // [B,EP_]
  int idx = blockIdx.x * blockDim.x + threadIdx.x;   // B*EP
  if (idx >= B_ * EP_) return;
  int b = idx / EP_;
  int k = idx - b * EP_;
  float v = (k < E_) ? table[(size_t)toks[b] * E_ + k] : 0.0f;
  X16[idx] = (_Float16)v;
}

// f32 -> f16 with column padding (row-major).
__global__ void conv_pad(const float* __restrict__ src, _Float16* __restrict__ dst,
                         int rows, int cols, int colsPad) {
  int idx = blockIdx.x * blockDim.x + threadIdx.x;
  if (idx >= rows * colsPad) return;
  int r = idx / colsPad;
  int k = idx - r * colsPad;
  dst[idx] = (k < cols) ? (_Float16)src[(size_t)r * cols + k] : (_Float16)0.0f;
}

__global__ void bias_sum(const float* __restrict__ a, const float* __restrict__ b,
                         float* __restrict__ d, int n) {
  int i = blockIdx.x * blockDim.x + threadIdx.x;
  if (i < n) d[i] = a[i] + b[i];
}

__global__ void zero_f32(float* p, int n) {
  int i = blockIdx.x * blockDim.x + threadIdx.x;
  if (i < n) p[i] = 0.0f;
}
__global__ void zero_f16(_Float16* p, int n) {
  int i = blockIdx.x * blockDim.x + threadIdx.x;
  if (i < n) p[i] = (_Float16)0.0f;
}
__global__ void init_tokens(const int* __restrict__ tgt0, int* __restrict__ cur) {
  int i = threadIdx.x;
  if (i < B_) cur[i] = tgt0[i];
}

// Per-batch-row argmax over V; then pick next token (teacher forcing select).
__global__ void argmax_next(const float* __restrict__ logits,   // [B,V] for this step
                            const int* __restrict__ tgt_row,    // target[t]
                            const int* __restrict__ tf,         // &tf_mask[t]
                            int* __restrict__ cur) {
  __shared__ float sv[256];
  __shared__ int   si[256];
  int b = blockIdx.x;
  const float* row = logits + (size_t)b * V_;
  float best = -3.4e38f; int bi = 0;
  for (int v = threadIdx.x; v < V_; v += 256) {
    float x = row[v];
    if (x > best || (x == best && v < bi)) { best = x; bi = v; }
  }
  sv[threadIdx.x] = best; si[threadIdx.x] = bi;
  __syncthreads();
  for (int s = 128; s > 0; s >>= 1) {
    if ((int)threadIdx.x < s) {
      float ov = sv[threadIdx.x + s]; int oi = si[threadIdx.x + s];
      if (ov > sv[threadIdx.x] || (ov == sv[threadIdx.x] && oi < si[threadIdx.x])) {
        sv[threadIdx.x] = ov; si[threadIdx.x] = oi;
      }
    }
    __syncthreads();
  }
  if (threadIdx.x == 0) cur[b] = (tf[0] > 0) ? tgt_row[b] : si[0];
}

// ---------------- host driver ----------------------------------------------

extern "C" void kernel_launch(void* const* d_in, const int* in_sizes, int n_in,
                              void* d_out, int out_size, void* d_ws, size_t ws_size,
                              hipStream_t stream) {
  (void)in_sizes; (void)n_in; (void)out_size; (void)ws_size;
  const int*   src       = (const int*)d_in[0];    // [S,B]
  const int*   target    = (const int*)d_in[1];    // [T,B]
  const int*   tf_mask   = (const int*)d_in[2];    // [T]
  const float* enc_embed = (const float*)d_in[3];  // [V,E]
  const float* dec_embed = (const float*)d_in[4];
  const float* eW_ih0 = (const float*)d_in[5];
  const float* eW_hh0 = (const float*)d_in[6];
  const float* eb_ih0 = (const float*)d_in[7];
  const float* eb_hh0 = (const float*)d_in[8];
  const float* eW_ih1 = (const float*)d_in[9];
  const float* eW_hh1 = (const float*)d_in[10];
  const float* eb_ih1 = (const float*)d_in[11];
  const float* eb_hh1 = (const float*)d_in[12];
  const float* dW_ih0 = (const float*)d_in[13];
  const float* dW_hh0 = (const float*)d_in[14];
  const float* db_ih0 = (const float*)d_in[15];
  const float* db_hh0 = (const float*)d_in[16];
  const float* dW_ih1 = (const float*)d_in[17];
  const float* dW_hh1 = (const float*)d_in[18];
  const float* db_ih1 = (const float*)d_in[19];
  const float* db_hh1 = (const float*)d_in[20];
  const float* fc_W   = (const float*)d_in[21];    // [V,H]
  const float* fc_b   = (const float*)d_in[22];    // [V]
  float* out = (float*)d_out;                      // [T,B,V]

  // workspace bump allocator (256B aligned)
  char* ws = (char*)d_ws;
  size_t off = 0;
  auto alloc = [&](size_t bytes) -> void* {
    void* p = ws + off;
    off = (off + bytes + 255) & ~(size_t)255;
    return p;
  };
  _Float16* Wih_e0 = (_Float16*)alloc((size_t)G4H_ * EP_ * 2);
  _Float16* Whh_e0 = (_Float16*)alloc((size_t)G4H_ * H_  * 2);
  _Float16* Wih_e1 = (_Float16*)alloc((size_t)G4H_ * H_  * 2);
  _Float16* Whh_e1 = (_Float16*)alloc((size_t)G4H_ * H_  * 2);
  _Float16* Wih_d0 = (_Float16*)alloc((size_t)G4H_ * EP_ * 2);
  _Float16* Whh_d0 = (_Float16*)alloc((size_t)G4H_ * H_  * 2);
  _Float16* Wih_d1 = (_Float16*)alloc((size_t)G4H_ * H_  * 2);
  _Float16* Whh_d1 = (_Float16*)alloc((size_t)G4H_ * H_  * 2);
  _Float16* fcW16  = (_Float16*)alloc((size_t)V_ * H_ * 2);      // 32.8 MB, L2-resident
  float* bias_e0 = (float*)alloc(G4H_ * 4);
  float* bias_e1 = (float*)alloc(G4H_ * 4);
  float* bias_d0 = (float*)alloc(G4H_ * 4);
  float* bias_d1 = (float*)alloc(G4H_ * 4);
  _Float16* X16   = (_Float16*)alloc((size_t)B_ * EP_ * 2);
  _Float16* H0_16 = (_Float16*)alloc((size_t)B_ * H_ * 2);
  _Float16* H1_16 = (_Float16*)alloc((size_t)B_ * H_ * 2);
  float* gates = (float*)alloc((size_t)B_ * G4H_ * 4);
  float* c0    = (float*)alloc((size_t)B_ * H_ * 4);
  float* c1    = (float*)alloc((size_t)B_ * H_ * 4);
  int*   cur   = (int*)alloc(B_ * 4);

  const int TB = 256;
  auto blks = [](int n, int tb) { return (n + tb - 1) / tb; };

  // ---- one-time conversions (per call; deterministic) ----
  conv_pad<<<blks(G4H_*EP_,TB),TB,0,stream>>>(eW_ih0, Wih_e0, G4H_, E_, EP_);
  conv_pad<<<blks(G4H_*H_, TB),TB,0,stream>>>(eW_hh0, Whh_e0, G4H_, H_, H_);
  conv_pad<<<blks(G4H_*H_, TB),TB,0,stream>>>(eW_ih1, Wih_e1, G4H_, H_, H_);
  conv_pad<<<blks(G4H_*H_, TB),TB,0,stream>>>(eW_hh1, Whh_e1, G4H_, H_, H_);
  conv_pad<<<blks(G4H_*EP_,TB),TB,0,stream>>>(dW_ih0, Wih_d0, G4H_, E_, EP_);
  conv_pad<<<blks(G4H_*H_, TB),TB,0,stream>>>(dW_hh0, Whh_d0, G4H_, H_, H_);
  conv_pad<<<blks(G4H_*H_, TB),TB,0,stream>>>(dW_ih1, Wih_d1, G4H_, H_, H_);
  conv_pad<<<blks(G4H_*H_, TB),TB,0,stream>>>(dW_hh1, Whh_d1, G4H_, H_, H_);
  conv_pad<<<blks(V_*H_,   TB),TB,0,stream>>>(fc_W,   fcW16,  V_,   H_, H_);
  bias_sum<<<blks(G4H_,TB),TB,0,stream>>>(eb_ih0, eb_hh0, bias_e0, G4H_);
  bias_sum<<<blks(G4H_,TB),TB,0,stream>>>(eb_ih1, eb_hh1, bias_e1, G4H_);
  bias_sum<<<blks(G4H_,TB),TB,0,stream>>>(db_ih0, db_hh0, bias_d0, G4H_);
  bias_sum<<<blks(G4H_,TB),TB,0,stream>>>(db_ih1, db_hh1, bias_d1, G4H_);

  // ---- state init ----
  zero_f32<<<blks(B_*H_,TB),TB,0,stream>>>(c0, B_*H_);
  zero_f32<<<blks(B_*H_,TB),TB,0,stream>>>(c1, B_*H_);
  zero_f16<<<blks(B_*H_,TB),TB,0,stream>>>(H0_16, B_*H_);
  zero_f16<<<blks(B_*H_,TB),TB,0,stream>>>(H1_16, B_*H_);
  zero_f32<<<blks(B_*V_,TB),TB,0,stream>>>(out, B_*V_);   // outputs[0] = 0

  // ---- encoder: S sequential steps ----
  for (int s = 0; s < S_; ++s) {
    embed_gather<<<blks(B_*EP_,TB),TB,0,stream>>>(src + s*B_, enc_embed, X16);
    lstm_gates_wmma<<<64,TB,0,stream>>>(X16, EP_, Wih_e0, H0_16, Whh_e0, gates);
    lstm_update<<<blks(B_*H_,TB),TB,0,stream>>>(gates, bias_e0, c0, H0_16);
    lstm_gates_wmma<<<64,TB,0,stream>>>(H0_16, H_, Wih_e1, H1_16, Whh_e1, gates);
    lstm_update<<<blks(B_*H_,TB),TB,0,stream>>>(gates, bias_e1, c1, H1_16);
  }

  // ---- decoder: 63 sequential steps ----
  init_tokens<<<1,64,0,stream>>>(target, cur);
  for (int t = 1; t < T_; ++t) {
    embed_gather<<<blks(B_*EP_,TB),TB,0,stream>>>(cur, dec_embed, X16);
    lstm_gates_wmma<<<64,TB,0,stream>>>(X16, EP_, Wih_d0, H0_16, Whh_d0, gates);
    lstm_update<<<blks(B_*H_,TB),TB,0,stream>>>(gates, bias_d0, c0, H0_16);
    lstm_gates_wmma<<<64,TB,0,stream>>>(H0_16, H_, Wih_d1, H1_16, Whh_d1, gates);
    lstm_update<<<blks(B_*H_,TB),TB,0,stream>>>(gates, bias_d1, c1, H1_16);
    float* step_out = out + (size_t)t * B_ * V_;
    fc_wmma<<<250,TB,0,stream>>>(H1_16, fcW16, fc_b, step_out);
    argmax_next<<<B_,TB,0,stream>>>(step_out, target + t*B_, tf_mask + t, cur);
  }
}
